// GPTDecoderModel_7825430413769
// MI455X (gfx1250) — matrix operand order repaired and checked by reading, not compiled, not run on hardware
//
#include <hip/hip_runtime.h>
#include <hip/hip_bf16.h>
#include <math.h>

// ---------------- model constants ----------------
#define D_MODEL 512
#define SEQ_T   256
#define NHEAD   8
#define HDK     64
#define NLAYER  6
#define DFF_N   2048
#define VOCAB   50257
#define BATCH   8
#define BT      (BATCH*SEQ_T)      // 2048 rows
#define VPAD    50304              // 786 * 64 (padded vocab rows for packed Wout)

typedef __attribute__((ext_vector_type(16))) _Float16 v16h;
typedef __attribute__((ext_vector_type(8)))  float    v8f;

union Frag { v16h v; uint4 q[2]; };

enum { GF_RELU=1, GF_F16=2, GF_F32=4, GF_CAUSAL=8, GF_NGUARD=16, GF_RES=32,
       GF_BIAS=64 };

struct GemmP {
  const _Float16* A; long sAo; long sAi; int lda;   // A: [M,K] f16 row-major
  const _Float16* B; long sBo; long sBi; int ldb;   // B: [N,K] f16 row-major (B^T)
  float* Cf; _Float16* Ch; long sCo; long sCi; int ldc;
  const float* bias;                                 // length N (when GF_BIAS)
  int M, N, K, inner;
  float alpha;
};

// ---------------------------------------------------------------------------
// WMMA GEMM: one wave -> 32x64 strip of C (2 M-tiles x 4 N-tiles), K in steps
// of 64 with double-buffered fragments (software pipelining).
// A-frag (16b A 16x32): lane m=l%16; halves j: K=(j%8)+8*(l/16)+16*(j/8)
//   -> two contiguous 16B loads per lane (offset 8*half and +16).
// B-frag (16b B 32x16): lane n=l%16; halves j: K=(j%8)+8*(l/16)+16*(j/8)
//   -> two contiguous 16B loads per lane from [N,K]-packed weights.
// C/D (f32 16x16): vgpr r, lanes0-15: M=r,N=l; lanes16-31: M=r+8,N=l-16.
// ---------------------------------------------------------------------------
template<int FLAGS>
__global__ __launch_bounds__(256) void gemm_wmma(GemmP p) {
  const int wave = threadIdx.x >> 5;
  const int lane = threadIdx.x & 31;
  const int z  = blockIdx.z;
  const int zo = z / p.inner, zi = z % p.inner;
  const _Float16* A = p.A + zo * p.sAo + zi * p.sAi;
  const _Float16* B = p.B + zo * p.sBo + zi * p.sBi;
  const long coff = zo * p.sCo + zi * p.sCi;

  const int m0 = (blockIdx.y * 8 + wave) * 32;
  if (m0 >= p.M) return;
  const int n0 = blockIdx.x * 64;
  if (FLAGS & GF_CAUSAL) {
    if (n0 > m0 + 31) return;          // strip entirely above the diagonal
  }

  const int half = lane >> 4;
  const int ln   = lane & 15;
  const int lda = p.lda, ldb = p.ldb;
  const _Float16* Ar0 = A + (long)(m0 + ln) * lda + 8 * half;        // M-tile 0
  const _Float16* Ar1 = Ar0 + (long)16 * lda;                        // M-tile 1
  const _Float16* Br  = B + (long)(n0 + ln) * ldb + 8 * half;

  Frag a0[2], b0[4], a1[2], b1[4];

  auto loadk = [&](int kb, Frag (&af)[2], Frag (&bf)[4]) {
    const _Float16* r0 = Ar0 + kb;
    af[0].q[0] = *(const uint4*)(r0);
    af[0].q[1] = *(const uint4*)(r0 + 16);
    const _Float16* r1 = Ar1 + kb;
    af[1].q[0] = *(const uint4*)(r1);
    af[1].q[1] = *(const uint4*)(r1 + 16);
#pragma unroll
    for (int t = 0; t < 4; ++t) {
      const _Float16* br = Br + (long)t * 16 * ldb + kb;
      bf[t].q[0] = *(const uint4*)(br);
      bf[t].q[1] = *(const uint4*)(br + 16);
    }
  };

  v8f acc[2][4] = {};
  auto mm8 = [&](Frag (&af)[2], Frag (&bf)[4]) {
#pragma unroll
    for (int mi = 0; mi < 2; ++mi)
#pragma unroll
      for (int t = 0; t < 4; ++t)
        acc[mi][t] = __builtin_amdgcn_wmma_f32_16x16x32_f16(
            false, af[mi].v, false, bf[t].v, (short)0, acc[mi][t], false, false);
  };

  // K is always a multiple of 64 here (64/256/512/2048).
  loadk(0, a0, b0);
  for (int kb = 0; kb < p.K; kb += 64) {
    loadk(kb + 32, a1, b1);
    mm8(a0, b0);
    if (kb + 64 < p.K) loadk(kb + 64, a0, b0);
    mm8(a1, b1);
  }

#pragma unroll
  for (int mi = 0; mi < 2; ++mi)
#pragma unroll
    for (int t = 0; t < 4; ++t) {
      const int n = n0 + t * 16 + ln;
      if ((FLAGS & GF_NGUARD) && n >= p.N) continue;
      const float bia = (FLAGS & GF_BIAS) ? p.bias[n] : 0.0f;
#pragma unroll
      for (int r = 0; r < 8; ++r) {
        const int mm = m0 + mi * 16 + half * 8 + r;
        float v = acc[mi][t][r] * p.alpha + bia;
        if (FLAGS & GF_RELU) v = fmaxf(v, 0.0f);
        const long ci = coff + (long)mm * p.ldc + n;
        if (FLAGS & GF_RES) v += p.Cf[ci];     // residual accumulate in place
        if (FLAGS & GF_F32) p.Cf[ci] = v;
        if (FLAGS & GF_F16) p.Ch[ci] = (_Float16)v;
      }
    }
}

// ---------------- embedding + sinusoidal positional encoding ----------------
__global__ __launch_bounds__(256) void embed_pe(const int* idx, const float* emb,
                                                float* x) {
  const long tid = (long)blockIdx.x * blockDim.x + threadIdx.x;
  if (tid >= (long)BT * D_MODEL) return;
  const int bt = (int)(tid / D_MODEL);
  const int d  = (int)(tid % D_MODEL);
  const int t  = bt % SEQ_T;
  const float freq = __expf(-((float)(d & ~1)) * (9.2103403719761836f / (float)D_MODEL));
  const float ang  = (float)t * freq;
  const float pe   = (d & 1) ? __cosf(ang) : __sinf(ang);
  x[tid] = emb[(long)idx[bt] * D_MODEL + d] + pe;
}

// ---------------- layernorm over D=512, f16 output for WMMA A ----------------
__global__ __launch_bounds__(256) void ln_kernel(const float* x, const float* g,
                                                 const float* b, _Float16* out) {
  __shared__ float r1[256], r2[256];
  const int row = blockIdx.x;
  const float* xr = x + (long)row * D_MODEL;
  const int tx = threadIdx.x;
  const float a0 = xr[tx], a1 = xr[tx + 256];
  r1[tx] = a0 + a1;
  r2[tx] = a0 * a0 + a1 * a1;
  __syncthreads();
  for (int st = 128; st > 0; st >>= 1) {
    if (tx < st) { r1[tx] += r1[tx + st]; r2[tx] += r2[tx + st]; }
    __syncthreads();
  }
  const float mean = r1[0] * (1.0f / D_MODEL);
  const float var  = r2[0] * (1.0f / D_MODEL) - mean * mean;
  const float inv  = rsqrtf(var + 1e-5f);
  _Float16* o = out + (long)row * D_MODEL;
  o[tx]       = (_Float16)((a0 - mean) * inv * g[tx]       + b[tx]);
  o[tx + 256] = (_Float16)((a1 - mean) * inv * g[tx + 256] + b[tx + 256]);
}

// ---------------- causal softmax: one block per (b,h,t) row, T==blockDim ----
__global__ __launch_bounds__(256) void softmax_kernel(const float* S, _Float16* P) {
  __shared__ float red[256];
  const int row = blockIdx.x;
  const int t   = row & (SEQ_T - 1);
  const int s   = threadIdx.x;
  const float v = (s <= t) ? S[(long)row * SEQ_T + s] : -INFINITY;
  red[s] = v; __syncthreads();
  for (int st = 128; st > 0; st >>= 1) {
    if (s < st) red[s] = fmaxf(red[s], red[s + st]);
    __syncthreads();
  }
  const float mx = red[0];
  __syncthreads();
  const float e = (s <= t) ? __expf(v - mx) : 0.0f;
  red[s] = e; __syncthreads();
  for (int st = 128; st > 0; st >>= 1) {
    if (s < st) red[s] += red[s + st];
    __syncthreads();
  }
  P[(long)row * SEQ_T + s] = (_Float16)(e / red[0]);
}

// ---------------- V [B,T,D] f16 -> Vt [B,H,64,T] f16 (for A*V^T GEMM) -------
__global__ __launch_bounds__(256) void trans_v(const _Float16* vh, _Float16* vt) {
  const long tid = (long)blockIdx.x * blockDim.x + threadIdx.x;
  if (tid >= (long)BT * D_MODEL) return;
  const int bt = (int)(tid / D_MODEL);
  const int e  = (int)(tid % D_MODEL);
  const int b = bt / SEQ_T, t = bt % SEQ_T;
  const int h = e >> 6, k = e & 63;
  vt[(((long)b * NHEAD + h) * HDK + k) * SEQ_T + t] = vh[tid];
}

// ---------------- weight packers (f32 -> f16, transposed [N,K]) -------------
// W: [K,N] f32  ->  out: [NPAD,K] f16, zero-filled for n>=N
__global__ __launch_bounds__(256) void pack_tr(const float* W, _Float16* o,
                                               int K, int N, int NPAD) {
  const long tid = (long)blockIdx.x * blockDim.x + threadIdx.x;
  if (tid >= (long)NPAD * K) return;
  const int n = (int)(tid / K);
  const int k = (int)(tid % K);
  o[tid] = (n < N) ? (_Float16)W[(long)k * N + n] : (_Float16)0.0f;
}

// Wq layer slice [H,D,DK] f32 -> out [H*DK, D] f16  (row e=h*64+k holds W[h,:,k])
__global__ __launch_bounds__(256) void pack_head(const float* W, _Float16* o) {
  const long tid = (long)blockIdx.x * blockDim.x + threadIdx.x;
  if (tid >= (long)D_MODEL * D_MODEL) return;
  const int e = (int)(tid / D_MODEL);
  const int d = (int)(tid % D_MODEL);
  o[tid] = (_Float16)W[((long)(e >> 6) * D_MODEL + d) * HDK + (e & 63)];
}

// ---------------------------------------------------------------------------
template<int FLAGS>
static void launch_gemm(const GemmP& p, int batches, hipStream_t stream) {
  dim3 grid((p.N + 63) / 64, (p.M + 255) / 256, batches);
  gemm_wmma<FLAGS><<<grid, 256, 0, stream>>>(p);
}

extern "C" void kernel_launch(void* const* d_in, const int* in_sizes, int n_in,
                              void* d_out, int out_size, void* d_ws, size_t ws_size,
                              hipStream_t stream) {
  (void)in_sizes; (void)n_in; (void)out_size; (void)ws_size;
  const int*   idx  = (const int*)  d_in[0];
  const float* emb  = (const float*)d_in[1];
  const float* Wq   = (const float*)d_in[2];
  const float* Wk   = (const float*)d_in[3];
  const float* Wv   = (const float*)d_in[4];
  const float* Wo   = (const float*)d_in[5];
  const float* ln1g = (const float*)d_in[6];
  const float* ln1b = (const float*)d_in[7];
  const float* ln2g = (const float*)d_in[8];
  const float* ln2b = (const float*)d_in[9];
  const float* W1   = (const float*)d_in[10];
  const float* b1   = (const float*)d_in[11];
  const float* W2   = (const float*)d_in[12];
  const float* b2   = (const float*)d_in[13];
  const float* lnfg = (const float*)d_in[14];
  const float* lnfb = (const float*)d_in[15];
  const float* Wout = (const float*)d_in[16];
  const float* bout = (const float*)d_in[17];
  float* out = (float*)d_out;

  // ---- workspace carve-up (256B aligned regions) ----
  char* ws = (char*)d_ws; size_t off = 0;
  auto alloc = [&](size_t bytes) -> void* {
    void* p = ws + off; off += (bytes + 255) & ~(size_t)255; return p;
  };
  float*    x    = (float*)   alloc((size_t)BT * D_MODEL * 4);
  _Float16* xnh  = (_Float16*)alloc((size_t)BT * D_MODEL * 2);
  _Float16* qh   = (_Float16*)alloc((size_t)BT * D_MODEL * 2);
  _Float16* kh   = (_Float16*)alloc((size_t)BT * D_MODEL * 2);
  _Float16* vh   = (_Float16*)alloc((size_t)BT * D_MODEL * 2);
  _Float16* vt   = (_Float16*)alloc((size_t)BT * D_MODEL * 2);
  _Float16* oh   = (_Float16*)alloc((size_t)BT * D_MODEL * 2);
  float*    S    = (float*)   alloc((size_t)BATCH * NHEAD * SEQ_T * SEQ_T * 4);
  _Float16* P    = (_Float16*)alloc((size_t)BATCH * NHEAD * SEQ_T * SEQ_T * 2);
  _Float16* ffh  = (_Float16*)alloc((size_t)BT * DFF_N * 2);
  _Float16* wqt  = (_Float16*)alloc((size_t)D_MODEL * D_MODEL * 2);
  _Float16* wkt  = (_Float16*)alloc((size_t)D_MODEL * D_MODEL * 2);
  _Float16* wvt  = (_Float16*)alloc((size_t)D_MODEL * D_MODEL * 2);
  _Float16* wot  = (_Float16*)alloc((size_t)D_MODEL * D_MODEL * 2);
  _Float16* w1t  = (_Float16*)alloc((size_t)DFF_N * D_MODEL * 2);
  _Float16* w2t  = (_Float16*)alloc((size_t)D_MODEL * DFF_N * 2);
  _Float16* wout = (_Float16*)alloc((size_t)VPAD * D_MODEL * 2);

  const float scl = 0.044194173824159216f;  // 1/sqrt(D_MODEL)
  const int TPB = 256;
  auto nblk = [](long n) { return (unsigned)((n + 255) / 256); };

  // pack vocab projection once (51.5 MB f16: resident in 192 MB L2, reused
  // across all 2048 output rows)
  pack_tr<<<nblk((long)VPAD * D_MODEL), TPB, 0, stream>>>(Wout, wout, D_MODEL, VOCAB, VPAD);

  // token embedding + positional encoding
  embed_pe<<<nblk((long)BT * D_MODEL), TPB, 0, stream>>>(idx, emb, x);

  for (int l = 0; l < NLAYER; ++l) {
    const long wqkv = (long)l * NHEAD * D_MODEL * HDK;
    pack_head<<<nblk((long)D_MODEL * D_MODEL), TPB, 0, stream>>>(Wq + wqkv, wqt);
    pack_head<<<nblk((long)D_MODEL * D_MODEL), TPB, 0, stream>>>(Wk + wqkv, wkt);
    pack_head<<<nblk((long)D_MODEL * D_MODEL), TPB, 0, stream>>>(Wv + wqkv, wvt);
    pack_tr<<<nblk((long)D_MODEL * D_MODEL), TPB, 0, stream>>>(
        Wo + (long)l * D_MODEL * D_MODEL, wot, D_MODEL, D_MODEL, D_MODEL);
    pack_tr<<<nblk((long)DFF_N * D_MODEL), TPB, 0, stream>>>(
        W1 + (long)l * D_MODEL * DFF_N, w1t, D_MODEL, DFF_N, DFF_N);
    pack_tr<<<nblk((long)D_MODEL * DFF_N), TPB, 0, stream>>>(
        W2 + (long)l * DFF_N * D_MODEL, w2t, DFF_N, D_MODEL, D_MODEL);

    // --- attention ---
    ln_kernel<<<BT, TPB, 0, stream>>>(x, ln1g + (long)l * D_MODEL,
                                      ln1b + (long)l * D_MODEL, xnh);
    GemmP pq = {xnh,0,0,D_MODEL, wqt,0,0,D_MODEL, nullptr,qh,0,0,D_MODEL,
                nullptr, BT,D_MODEL,D_MODEL,1, 1.0f};
    launch_gemm<GF_F16>(pq, 1, stream);
    GemmP pk = pq; pk.B = wkt; pk.Ch = kh; launch_gemm<GF_F16>(pk, 1, stream);
    GemmP pv = pq; pv.B = wvt; pv.Ch = vh; launch_gemm<GF_F16>(pv, 1, stream);
    trans_v<<<nblk((long)BT * D_MODEL), TPB, 0, stream>>>(vh, vt);

    // scores S[b,h,t,s] = scl * Q_h K_h^T  (strips above diagonal skipped)
    GemmP ps = {qh,(long)SEQ_T*D_MODEL,HDK,D_MODEL,
                kh,(long)SEQ_T*D_MODEL,HDK,D_MODEL,
                S,nullptr,(long)NHEAD*SEQ_T*SEQ_T,(long)SEQ_T*SEQ_T,SEQ_T,
                nullptr, SEQ_T,SEQ_T,HDK,NHEAD, scl};
    launch_gemm<GF_F32 | GF_CAUSAL>(ps, BATCH * NHEAD, stream);
    softmax_kernel<<<BATCH * NHEAD * SEQ_T, TPB, 0, stream>>>(S, P);

    // O_h = P V_h  -> packed heads f16 [BT, D]
    GemmP pav = {P,(long)NHEAD*SEQ_T*SEQ_T,(long)SEQ_T*SEQ_T,SEQ_T,
                 vt,(long)NHEAD*HDK*SEQ_T,(long)HDK*SEQ_T,SEQ_T,
                 nullptr,oh,(long)SEQ_T*D_MODEL,HDK,D_MODEL,
                 nullptr, SEQ_T,HDK,SEQ_T,NHEAD, 1.0f};
    launch_gemm<GF_F16>(pav, BATCH * NHEAD, stream);

    // x += O Wo
    GemmP pp = {oh,0,0,D_MODEL, wot,0,0,D_MODEL, x,nullptr,0,0,D_MODEL,
                nullptr, BT,D_MODEL,D_MODEL,1, 1.0f};
    launch_gemm<GF_F32 | GF_RES>(pp, 1, stream);

    // --- FFN ---
    ln_kernel<<<BT, TPB, 0, stream>>>(x, ln2g + (long)l * D_MODEL,
                                      ln2b + (long)l * D_MODEL, xnh);
    GemmP pf1 = {xnh,0,0,D_MODEL, w1t,0,0,D_MODEL, nullptr,ffh,0,0,DFF_N,
                 b1 + (long)l * DFF_N, BT,DFF_N,D_MODEL,1, 1.0f};
    launch_gemm<GF_F16 | GF_RELU | GF_BIAS>(pf1, 1, stream);
    GemmP pf2 = {ffh,0,0,DFF_N, w2t,0,0,DFF_N, x,nullptr,0,0,D_MODEL,
                 b2 + (long)l * D_MODEL, BT,D_MODEL,DFF_N,1, 1.0f};
    launch_gemm<GF_F32 | GF_RES | GF_BIAS>(pf2, 1, stream);
  }

  // final LN + vocab projection
  ln_kernel<<<BT, TPB, 0, stream>>>(x, lnfg, lnfb, xnh);
  GemmP pl = {xnh,0,0,D_MODEL, wout,0,0,D_MODEL, out,nullptr,0,0,VOCAB,
              bout, BT,VOCAB,D_MODEL,1, 1.0f};
  launch_gemm<GF_F32 | GF_NGUARD | GF_BIAS>(pl, 1, stream);
}
